// MelGraphSAGE_29583734734919
// MI455X (gfx1250) — compile-verified, matrix operand order checked by reading, not run on hardware
//
#include <hip/hip_runtime.h>

// ---------------------------------------------------------------------------
// MelGraphSAGE for MI455X (gfx1250, wave32, WMMA bf16 16x16x32)
// ---------------------------------------------------------------------------

#define N_NODES 20000
#define N_EDGES 320000
#define IN_DIM  8192
#define HID     256
#define NCLS    4

typedef __attribute__((ext_vector_type(16))) __bf16 v16bf;
typedef __attribute__((ext_vector_type(8)))  __bf16 v8bf;
typedef __attribute__((ext_vector_type(8)))  float  v8f;

__device__ __forceinline__ v8f wmma_bf16(v16bf a, v16bf b, v8f c) {
  // (neg_a, A, neg_b, B, c_mod, C, reuse_a, reuse_b)
  return __builtin_amdgcn_wmma_f32_16x16x32_bf16(false, a, false, b, (short)0, c,
                                                 false, false);
}

// Build a 16x32 bf16 A-fragment for this lane from row-major f32 memory.
// `base` must already include  row*ld + k + kOff  (kOff = (lane>>4)*8).
// Fragment elems 0..7  = K kOff+0..7 ; elems 8..15 = K kOff+16..23.
__device__ __forceinline__ v16bf load_frag_f32(const float* __restrict__ base) {
  float4 a0 = *(const float4*)(base + 0);
  float4 a1 = *(const float4*)(base + 4);
  float4 a2 = *(const float4*)(base + 16);
  float4 a3 = *(const float4*)(base + 20);
  v16bf f;
  f[0]  = (__bf16)a0.x; f[1]  = (__bf16)a0.y; f[2]  = (__bf16)a0.z; f[3]  = (__bf16)a0.w;
  f[4]  = (__bf16)a1.x; f[5]  = (__bf16)a1.y; f[6]  = (__bf16)a1.z; f[7]  = (__bf16)a1.w;
  f[8]  = (__bf16)a2.x; f[9]  = (__bf16)a2.y; f[10] = (__bf16)a2.z; f[11] = (__bf16)a2.w;
  f[12] = (__bf16)a3.x; f[13] = (__bf16)a3.y; f[14] = (__bf16)a3.z; f[15] = (__bf16)a3.w;
  return f;
}

// B-fragment (16x32, column per lane) from pre-transposed bf16 W^T:
// base = Wt + n*ldk + k + kOff ; two contiguous 16B loads.
__device__ __forceinline__ v16bf load_frag_bf16(const __bf16* __restrict__ base) {
  v8bf b0 = *(const v8bf*)(base + 0);
  v8bf b1 = *(const v8bf*)(base + 16);
  return __builtin_shufflevector(b0, b1, 0, 1, 2, 3, 4, 5, 6, 7,
                                 8, 9, 10, 11, 12, 13, 14, 15);
}

// ---------------------------------------------------------------------------
// Prep: f32 [K][N] -> bf16 transposed [N][K]  (write-coalesced)
// ---------------------------------------------------------------------------
__global__ void k_transpose_bf16(const float* __restrict__ W,
                                 __bf16* __restrict__ Wt, int K, int N) {
  long long idx = (long long)blockIdx.x * blockDim.x + threadIdx.x;
  long long total = (long long)K * N;
  if (idx >= total) return;
  int n = (int)(idx / K);
  int k = (int)(idx % K);
  Wt[idx] = (__bf16)W[(long long)k * N + n];
}

__global__ void k_zero_f32(float* __restrict__ p, long long n) {
  long long idx = (long long)blockIdx.x * blockDim.x + threadIdx.x;
  if (idx < n) p[idx] = 0.0f;
}

// ---------------------------------------------------------------------------
// GEMM1: h1 = relu(x[20000x8192] @ W_pre[8192x256] + b_pre)
// Block = 8 waves = 4 row-tiles x 2 col-halves. Wave tile = 16 rows x 128 cols.
// ---------------------------------------------------------------------------
__global__ __launch_bounds__(256) void k_gemm_pre(const float* __restrict__ x,
                                                  const __bf16* __restrict__ Wt,
                                                  const float* __restrict__ bias,
                                                  float* __restrict__ h1) {
  const int lane = threadIdx.x & 31;
  const int wave = threadIdx.x >> 5;
  const int rowTile = blockIdx.x * 4 + (wave >> 1);
  if (rowTile * 16 >= N_NODES) return;
  const int row0    = rowTile * 16;
  const int colBase = (wave & 1) * 128;

  const int l15  = lane & 15;
  const int hi   = lane >> 4;
  const int kOff = hi * 8;

  v8f acc[8];
#pragma unroll
  for (int t = 0; t < 8; ++t) acc[t] = (v8f){0.f, 0.f, 0.f, 0.f, 0.f, 0.f, 0.f, 0.f};

  const float*  aBase = x  + (size_t)(row0 + l15) * IN_DIM + kOff;
  const __bf16* bBase = Wt + (size_t)(colBase + l15) * IN_DIM + kOff;

  for (int k = 0; k < IN_DIM; k += 32) {
    v16bf afrag = load_frag_f32(aBase + k);
#pragma unroll
    for (int t = 0; t < 8; ++t) {
      v16bf bfrag = load_frag_bf16(bBase + (size_t)t * 16 * IN_DIM + k);
      acc[t] = wmma_bf16(afrag, bfrag, acc[t]);
    }
  }

  // Epilogue: C layout — lanes 0-15: (M=v, N=lane); lanes 16-31: (M=v+8, N=lane-16)
#pragma unroll
  for (int t = 0; t < 8; ++t) {
    const int col = colBase + t * 16 + l15;
    const float b = bias[col];
#pragma unroll
    for (int v = 0; v < 8; ++v) {
      const int r = row0 + v + hi * 8;
      h1[(size_t)r * HID + col] = fmaxf(acc[t][v] + b, 0.0f);
    }
  }
}

// ---------------------------------------------------------------------------
// Edge scatter: sum[dst] += h[src]  (+ degree count on first pass)
// One edge per 256-thread block; lane = feature column. L2-resident atomics.
// ---------------------------------------------------------------------------
__global__ __launch_bounds__(256) void k_scatter(const float* __restrict__ h,
                                                 const int* __restrict__ src,
                                                 const int* __restrict__ dst,
                                                 float* __restrict__ sum,
                                                 float* __restrict__ cnt,
                                                 int addCnt) {
  const int e = blockIdx.x;
  const int s = src[e];
  const int d = dst[e];
  atomicAdd(&sum[(size_t)d * HID + threadIdx.x], h[(size_t)s * HID + threadIdx.x]);
  if (addCnt && threadIdx.x == 0) atomicAdd(&cnt[d], 1.0f);
}

// In-place:  sum[n][c] /= max(cnt[n], 1)
__global__ void k_meanize(float* __restrict__ sum, const float* __restrict__ cnt) {
  long long idx = (long long)blockIdx.x * blockDim.x + threadIdx.x;
  if (idx >= (long long)N_NODES * HID) return;
  const int n = (int)(idx >> 8);
  sum[idx] *= 1.0f / fmaxf(cnt[n], 1.0f);
}

// ---------------------------------------------------------------------------
// SAGE GEMM: h2 = relu(mean @ W1_l + b1 + h1 @ W1_r)   (K = 256, fused dual-A)
// ---------------------------------------------------------------------------
__global__ __launch_bounds__(256) void k_gemm_sage(const float* __restrict__ mean,
                                                   const float* __restrict__ h,
                                                   const __bf16* __restrict__ WlT,
                                                   const __bf16* __restrict__ WrT,
                                                   const float* __restrict__ bias,
                                                   float* __restrict__ out) {
  const int lane = threadIdx.x & 31;
  const int wave = threadIdx.x >> 5;
  const int rowTile = blockIdx.x * 4 + (wave >> 1);
  if (rowTile * 16 >= N_NODES) return;
  const int row0    = rowTile * 16;
  const int colBase = (wave & 1) * 128;

  const int l15  = lane & 15;
  const int hi   = lane >> 4;
  const int kOff = hi * 8;

  v8f acc[8];
#pragma unroll
  for (int t = 0; t < 8; ++t) acc[t] = (v8f){0.f, 0.f, 0.f, 0.f, 0.f, 0.f, 0.f, 0.f};

  const float*  mBase  = mean + (size_t)(row0 + l15) * HID + kOff;
  const float*  hBase  = h    + (size_t)(row0 + l15) * HID + kOff;
  const __bf16* blBase = WlT  + (size_t)(colBase + l15) * HID + kOff;
  const __bf16* brBase = WrT  + (size_t)(colBase + l15) * HID + kOff;

#pragma unroll
  for (int k = 0; k < HID; k += 32) {
    v16bf am = load_frag_f32(mBase + k);
    v16bf ah = load_frag_f32(hBase + k);
#pragma unroll
    for (int t = 0; t < 8; ++t) {
      v16bf bl = load_frag_bf16(blBase + (size_t)t * 16 * HID + k);
      v16bf br = load_frag_bf16(brBase + (size_t)t * 16 * HID + k);
      acc[t] = wmma_bf16(am, bl, acc[t]);
      acc[t] = wmma_bf16(ah, br, acc[t]);
    }
  }

#pragma unroll
  for (int t = 0; t < 8; ++t) {
    const int col = colBase + t * 16 + l15;
    const float b = bias[col];
#pragma unroll
    for (int v = 0; v < 8; ++v) {
      const int r = row0 + v + hi * 8;
      out[(size_t)r * HID + col] = fmaxf(acc[t][v] + b, 0.0f);
    }
  }
}

// ---------------------------------------------------------------------------
// Final layer (N=4, scalar): out = mean2 @ W2_l + b2 + h2 @ W2_r
// ---------------------------------------------------------------------------
__global__ void k_final(const float* __restrict__ sum2, const float* __restrict__ cnt,
                        const float* __restrict__ h2,  const float* __restrict__ W2l,
                        const float* __restrict__ b2,  const float* __restrict__ W2r,
                        float* __restrict__ out) {
  int idx = blockIdx.x * blockDim.x + threadIdx.x;
  if (idx >= N_NODES * NCLS) return;
  const int n = idx >> 2;
  const int c = idx & 3;
  const float rs = 1.0f / fmaxf(cnt[n], 1.0f);
  const float* s  = sum2 + (size_t)n * HID;
  const float* hh = h2   + (size_t)n * HID;
  float acc = b2[c];
#pragma unroll 4
  for (int k = 0; k < HID; ++k)
    acc += s[k] * rs * W2l[k * NCLS + c] + hh[k] * W2r[k * NCLS + c];
  out[idx] = acc;
}

// ---------------------------------------------------------------------------
// Host orchestration
// ---------------------------------------------------------------------------
extern "C" void kernel_launch(void* const* d_in, const int* in_sizes, int n_in,
                              void* d_out, int out_size, void* d_ws, size_t ws_size,
                              hipStream_t stream) {
  const float* x     = (const float*)d_in[0];
  const int*   edges = (const int*)d_in[1];      // [2][E]: row0=src, row1=dst
  const float* W_pre = (const float*)d_in[2];
  const float* b_pre = (const float*)d_in[3];
  const float* W1_l  = (const float*)d_in[4];
  const float* b1_l  = (const float*)d_in[5];
  const float* W1_r  = (const float*)d_in[6];
  const float* W2_l  = (const float*)d_in[7];
  const float* b2_l  = (const float*)d_in[8];
  const float* W2_r  = (const float*)d_in[9];
  float* out = (float*)d_out;

  const int* src = edges;
  const int* dst = edges + N_EDGES;

  // ---- workspace layout (256B aligned slices) ----
  char* ws = (char*)d_ws;
  size_t off = 0;
  auto alloc = [&](size_t bytes) {
    void* p = ws + off;
    off += (bytes + 255) & ~(size_t)255;
    return p;
  };
  float*  h1    = (float*)alloc((size_t)N_NODES * HID * sizeof(float));   // 20.5 MB
  float*  h2    = (float*)alloc((size_t)N_NODES * HID * sizeof(float));   // 20.5 MB
  float*  sum   = (float*)alloc((size_t)N_NODES * HID * sizeof(float));   // 20.5 MB
  float*  cnt   = (float*)alloc((size_t)N_NODES * sizeof(float));
  __bf16* WtPre = (__bf16*)alloc((size_t)IN_DIM * HID * sizeof(__bf16));  // 4.2 MB
  __bf16* W1lT  = (__bf16*)alloc((size_t)HID * HID * sizeof(__bf16));
  __bf16* W1rT  = (__bf16*)alloc((size_t)HID * HID * sizeof(__bf16));

  // ---- 1. weight transposes / bf16 conversion ----
  {
    long long n = (long long)IN_DIM * HID;
    k_transpose_bf16<<<(unsigned)((n + 255) / 256), 256, 0, stream>>>(W_pre, WtPre, IN_DIM, HID);
  }
  {
    long long n = (long long)HID * HID;
    unsigned g = (unsigned)((n + 255) / 256);
    k_transpose_bf16<<<g, 256, 0, stream>>>(W1_l, W1lT, HID, HID);
    k_transpose_bf16<<<g, 256, 0, stream>>>(W1_r, W1rT, HID, HID);
  }

  // ---- 2. h1 = relu(x @ W_pre + b_pre) ----
  {
    unsigned rowTiles = N_NODES / 16;                 // 1250 (exact)
    unsigned grid = (rowTiles + 3) / 4;               // 4 row-tiles per block
    k_gemm_pre<<<grid, 256, 0, stream>>>(x, WtPre, b_pre, h1);
  }

  // ---- 3. aggregation 1: sum/cnt ----
  {
    long long n = (long long)N_NODES * HID;
    k_zero_f32<<<(unsigned)((n + 255) / 256), 256, 0, stream>>>(sum, n);
    k_zero_f32<<<(N_NODES + 255) / 256, 256, 0, stream>>>(cnt, N_NODES);
    k_scatter<<<N_EDGES, 256, 0, stream>>>(h1, src, dst, sum, cnt, 1);
    k_meanize<<<(unsigned)((n + 255) / 256), 256, 0, stream>>>(sum, cnt);
  }

  // ---- 4. h2 = relu(mean @ W1_l + b1 + h1 @ W1_r) ----
  {
    unsigned grid = (N_NODES / 16 + 3) / 4;
    k_gemm_sage<<<grid, 256, 0, stream>>>(sum, h1, W1lT, W1rT, b1_l, h2);
  }

  // ---- 5. aggregation 2 (reuse sum, reuse cnt) ----
  {
    long long n = (long long)N_NODES * HID;
    k_zero_f32<<<(unsigned)((n + 255) / 256), 256, 0, stream>>>(sum, n);
    k_scatter<<<N_EDGES, 256, 0, stream>>>(h2, src, dst, sum, cnt, 0);
  }

  // ---- 6. out = mean2 @ W2_l + b2 + h2 @ W2_r ----
  {
    unsigned grid = (N_NODES * NCLS + 255) / 256;
    k_final<<<grid, 256, 0, stream>>>(sum, cnt, h2, W2_l, b2_l, W2_r, out);
  }
}
